// DeepHamCritic_10934986736350
// MI455X (gfx1250) — compile-verified, matrix operand order checked by reading, not run on hardware
//
#include <hip/hip_runtime.h>
#include <hip/hip_bf16.h>

// ---------------------------------------------------------------------------
// Sizes (from reference)
// ---------------------------------------------------------------------------
static constexpr int N_NODES = 1024;
static constexpr int D_IN    = 128;
static constexpr int D_H     = 512;
static constexpr int N_EDGES = 32768;
static constexpr int N_ITEMS = N_EDGES + N_NODES;   // edges + appended self loops

typedef __attribute__((ext_vector_type(16))) _Float16 v16h;
typedef __attribute__((ext_vector_type(8)))  float    v8f;

// ---------------------------------------------------------------------------
// GEMM: Y[N x M] = X[N x K] @ W[K x M] + bias,  M = 512, K in {128, 512}.
// Block = 256 threads (8 waves) -> 64x64 output tile.
// Per 32-wide K-step: stage X tile (64x32) and W tile transposed (64 cols x 32 k)
// in LDS as f16 (float4-coalesced global reads), then each wave runs two
// v_wmma_f32_16x16x32_f16 on a shared A fragment (16x32 sub-tile per wave).
// A/B fragment = two contiguous 8xf16 (16 B) LDS runs per the ISA layout:
//   lane half 0 -> K {0..7, 16..23},  half 1 -> K {8..15, 24..31}.
// ---------------------------------------------------------------------------
__global__ void __launch_bounds__(256)
gemm_bias_wmma(const float* __restrict__ X, const float* __restrict__ W,
               const float* __restrict__ bias, float* __restrict__ Y,
               int K, int M) {
  __shared__ __align__(16) _Float16 Xs[64 * 32];    // [row][k]
  __shared__ __align__(16) _Float16 WsT[64 * 32];   // [col][k]  (transposed W)

  const int t     = threadIdx.x;
  const int lane  = t & 31;
  const int wave  = t >> 5;
  const int half  = lane >> 4;                 // K-half selector
  const int mr    = lane & 15;                 // A row / B col / D col
  const int rT    = (wave >> 1) * 16;          // wave row sub-tile: 0,16,32,48
  const int cT    = (wave & 1) * 32;           // wave col sub-tile: 0,32
  const int rBase = blockIdx.y * 64;
  const int cBase = blockIdx.x * 64;

  v8f acc0 = {}, acc1 = {};

  for (int kb = 0; kb < K; kb += 32) {
    __syncthreads();                           // protect LDS from prior step
    // ---- stage X tile: 64 rows x 32 k, float4 loads, cvt to f16 ----
#pragma unroll
    for (int i = 0; i < 2; ++i) {
      const int f   = t + i * 256;             // 0..511 float4 slots
      const int row = f >> 3;
      const int k4  = (f & 7) * 4;
      const float4 v = *(const float4*)&X[(size_t)(rBase + row) * K + kb + k4];
      _Float16* dst = &Xs[row * 32 + k4];
      dst[0] = (_Float16)v.x; dst[1] = (_Float16)v.y;
      dst[2] = (_Float16)v.z; dst[3] = (_Float16)v.w;
    }
    // ---- stage W tile transposed: WsT[col][k], float4 row reads ----
#pragma unroll
    for (int i = 0; i < 2; ++i) {
      const int f  = t + i * 256;              // 32 k-rows x 16 float4
      const int k  = f >> 4;
      const int c4 = (f & 15) * 4;
      const float4 v = *(const float4*)&W[(size_t)(kb + k) * M + cBase + c4];
      WsT[(c4 + 0) * 32 + k] = (_Float16)v.x;
      WsT[(c4 + 1) * 32 + k] = (_Float16)v.y;
      WsT[(c4 + 2) * 32 + k] = (_Float16)v.z;
      WsT[(c4 + 3) * 32 + k] = (_Float16)v.w;
    }
    __syncthreads();

    // ---- fragments from LDS (two aligned 16B runs each) ----
    v16h a, b0, b1;
    const _Float16* ap  = &Xs[(rT + mr) * 32 + half * 8];
    const _Float16* bp0 = &WsT[(cT + mr) * 32 + half * 8];
    const _Float16* bp1 = &WsT[(cT + 16 + mr) * 32 + half * 8];
#pragma unroll
    for (int e = 0; e < 8; ++e) {
      a[e]  = ap[e];   a[e + 8]  = ap[16 + e];
      b0[e] = bp0[e];  b0[e + 8] = bp0[16 + e];
      b1[e] = bp1[e];  b1[e + 8] = bp1[16 + e];
    }
    acc0 = __builtin_amdgcn_wmma_f32_16x16x32_f16(false, a, false, b0,
                                                  (short)0, acc0, false, false);
    acc1 = __builtin_amdgcn_wmma_f32_16x16x32_f16(false, a, false, b1,
                                                  (short)0, acc1, false, false);
  }

#pragma unroll
  for (int j = 0; j < 8; ++j) {
    const int row = rBase + rT + half * 8 + j;
    const int c0  = cBase + cT + mr;
    Y[(size_t)row * M + c0]      = acc0[j] + bias[c0];
    Y[(size_t)row * M + c0 + 16] = acc1[j] + bias[c0 + 16];
  }
}

// ---------------------------------------------------------------------------
// Orderable-uint encoding of float for atomic max (monotone in f)
// ---------------------------------------------------------------------------
__device__ __forceinline__ unsigned enc_f32(float f) {
  unsigned u = __float_as_uint(f);
  return (u & 0x80000000u) ? ~u : (u | 0x80000000u);
}
__device__ __forceinline__ float dec_f32(unsigned u) {
  u = (u & 0x80000000u) ? (u ^ 0x80000000u) : ~u;
  return __uint_as_float(u);
}

// ---------------------------------------------------------------------------
// Pass A: edge scores  s = att . leaky_relu(xl[src]+xr[dst], 0.2)
//         (src==dst real edges get -1e30), plus atomic segment-max into mbits.
// One wave per item; lane handles 16 coalesced dims.
// ---------------------------------------------------------------------------
__global__ void __launch_bounds__(256)
edge_score(const int* __restrict__ ei, const float* __restrict__ xl,
           const float* __restrict__ xr, const float* __restrict__ att,
           float* __restrict__ scores, unsigned* __restrict__ mbits) {
  const int item = blockIdx.x * (blockDim.x >> 5) + (threadIdx.x >> 5);
  const int lane = threadIdx.x & 31;
  if (item >= N_ITEMS) return;
  int s, d;
  if (item < N_EDGES) { s = ei[item]; d = ei[N_EDGES + item]; }
  else                { s = d = item - N_EDGES; }

  const float* pl = xl + (size_t)s * D_H;
  const float* pr = xr + (size_t)d * D_H;
  float p = 0.f;
#pragma unroll
  for (int j = 0; j < 16; ++j) {
    const int c = lane + 32 * j;
    float v = pl[c] + pr[c];
    v = (v > 0.f) ? v : 0.2f * v;
    p += att[c] * v;
  }
  for (int off = 16; off; off >>= 1) p += __shfl_xor(p, off, 32);
  if (item < N_EDGES && s == d) p = -1e30f;   // remove_self_loops
  if (lane == 0) {
    scores[item] = p;
    atomicMax(&mbits[d], enc_f32(p));
  }
}

// ---------------------------------------------------------------------------
// Pass B: a = exp(s - m[dst]); denom[dst] += a
// ---------------------------------------------------------------------------
__global__ void __launch_bounds__(256)
edge_exp(const int* __restrict__ ei, const float* __restrict__ scores,
         const unsigned* __restrict__ mbits, float* __restrict__ expa,
         float* __restrict__ denom) {
  const int item = blockIdx.x * blockDim.x + threadIdx.x;
  if (item >= N_ITEMS) return;
  const int d = (item < N_EDGES) ? ei[N_EDGES + item] : item - N_EDGES;
  const float a = expf(scores[item] - dec_f32(mbits[d]));
  expa[item] = a;
  atomicAdd(&denom[d], a);
}

// ---------------------------------------------------------------------------
// Pass C: agg[dst] += (a/denom[dst]) * xl[src]   (wave per item)
// ---------------------------------------------------------------------------
__global__ void __launch_bounds__(256)
edge_aggregate(const int* __restrict__ ei, const float* __restrict__ xl,
               const float* __restrict__ expa, const float* __restrict__ denom,
               float* __restrict__ agg) {
  const int item = blockIdx.x * (blockDim.x >> 5) + (threadIdx.x >> 5);
  const int lane = threadIdx.x & 31;
  if (item >= N_ITEMS) return;
  int s, d;
  if (item < N_EDGES) { s = ei[item]; d = ei[N_EDGES + item]; }
  else                { s = d = item - N_EDGES; }
  const float alpha = expa[item] / denom[d];
  const float* xs = xl + (size_t)s * D_H;
  float* od = agg + (size_t)d * D_H;
#pragma unroll
  for (int j = 0; j < 16; ++j) {
    const int c = lane + 32 * j;
    atomicAdd(&od[c], alpha * xs[c]);
  }
}

// ---------------------------------------------------------------------------
// Finalize layer: h = tanh(agg + bo)
// ---------------------------------------------------------------------------
__global__ void __launch_bounds__(256)
node_finalize(const float* __restrict__ agg, const float* __restrict__ bo,
              float* __restrict__ hout) {
  const int idx = blockIdx.x * blockDim.x + threadIdx.x;   // N * D_H
  hout[idx] = tanhf(agg[idx] + bo[idx & (D_H - 1)]);
}

// ---------------------------------------------------------------------------
// MLP head stage 1 (the 1 GiB stream): partial h1acc[j] += v[r]*Wm1[r][j]
// Block owns 256 rows; v-slice staged in LDS; coalesced float2 weight reads.
// ---------------------------------------------------------------------------
__global__ void __launch_bounds__(256)
mlp1_partial(const float* __restrict__ v, const float* __restrict__ Wm1,
             float* __restrict__ h1acc) {
  __shared__ float vs[256];
  const int t  = threadIdx.x;
  const int r0 = blockIdx.x * 256;
  vs[t] = v[r0 + t];
  __syncthreads();
  float acc0 = 0.f, acc1 = 0.f;
  for (int i = 0; i < 256; ++i) {
    const float vr = vs[i];
    const float2 w = ((const float2*)(Wm1 + (size_t)(r0 + i) * D_H))[t];
    acc0 += vr * w.x;
    acc1 += vr * w.y;
  }
  atomicAdd(&h1acc[2 * t],     acc0);
  atomicAdd(&h1acc[2 * t + 1], acc1);
}

// ---------------------------------------------------------------------------
// MLP stage 2: h2 = lrelu( lrelu(h1acc+bm1) @ Wm2 + bm2 ), 2 blocks x 256
// ---------------------------------------------------------------------------
__global__ void __launch_bounds__(256)
mlp2(const float* __restrict__ h1acc, const float* __restrict__ bm1,
     const float* __restrict__ Wm2, const float* __restrict__ bm2,
     float* __restrict__ h2) {
  __shared__ float h1s[D_H];
  const int t = threadIdx.x;
  for (int i = t; i < D_H; i += 256) {
    float v = h1acc[i] + bm1[i];
    h1s[i] = (v > 0.f) ? v : 0.01f * v;
  }
  __syncthreads();
  const int j = blockIdx.x * 256 + t;
  float acc = 0.f;
  for (int i = 0; i < D_H; ++i) acc += h1s[i] * Wm2[(size_t)i * D_H + j];
  acc += bm2[j];
  h2[j] = (acc > 0.f) ? acc : 0.01f * acc;
}

// ---------------------------------------------------------------------------
// MLP stage 3: out[0] = h2 . Wm3 + bm3
// ---------------------------------------------------------------------------
__global__ void __launch_bounds__(256)
mlp3(const float* __restrict__ h2, const float* __restrict__ Wm3,
     const float* __restrict__ bm3, float* __restrict__ out) {
  __shared__ float red[256];
  const int t = threadIdx.x;
  red[t] = h2[t] * Wm3[t] + h2[t + 256] * Wm3[t + 256];
  __syncthreads();
  for (int s = 128; s > 0; s >>= 1) {
    if (t < s) red[t] += red[t + s];
    __syncthreads();
  }
  if (t == 0) out[0] = red[0] + bm3[0];
}

// ---------------------------------------------------------------------------
// Host driver
// ---------------------------------------------------------------------------
static void run_gat_layer(const float* xin, int K, const int* ei,
                          const float* Wl, const float* bl,
                          const float* Wr, const float* br,
                          const float* att, const float* bo,
                          float* xl, float* xr, float* agg,
                          float* scores, float* expa,
                          unsigned* mbits, float* denom,
                          float* hout, hipStream_t stream) {
  dim3 gGemm(D_H / 64, N_NODES / 64);
  gemm_bias_wmma<<<gGemm, 256, 0, stream>>>(xin, Wl, bl, xl, K, D_H);
  gemm_bias_wmma<<<gGemm, 256, 0, stream>>>(xin, Wr, br, xr, K, D_H);

  hipMemsetAsync(mbits, 0, N_NODES * sizeof(unsigned), stream);       // == ordered -inf
  hipMemsetAsync(denom, 0, N_NODES * sizeof(float), stream);
  hipMemsetAsync(agg,   0, (size_t)N_NODES * D_H * sizeof(float), stream);

  const int waveBlocks = (N_ITEMS + 7) / 8;                           // 8 waves/block
  edge_score<<<waveBlocks, 256, 0, stream>>>(ei, xl, xr, att, scores, mbits);
  edge_exp<<<(N_ITEMS + 255) / 256, 256, 0, stream>>>(ei, scores, mbits, expa, denom);
  edge_aggregate<<<waveBlocks, 256, 0, stream>>>(ei, xl, expa, denom, agg);
  node_finalize<<<(N_NODES * D_H) / 256, 256, 0, stream>>>(agg, bo, hout);
}

extern "C" void kernel_launch(void* const* d_in, const int* in_sizes, int n_in,
                              void* d_out, int out_size, void* d_ws, size_t ws_size,
                              hipStream_t stream) {
  (void)in_sizes; (void)n_in; (void)out_size; (void)ws_size;

  const float* x   = (const float*)d_in[0];
  const int*   ei  = (const int*)  d_in[1];
  // Per-layer weights: [Wl, bl, Wr, br, att, bo] at indices 2.., 8.., 14..
  const float* Wm1 = (const float*)d_in[20];
  const float* bm1 = (const float*)d_in[21];
  const float* Wm2 = (const float*)d_in[22];
  const float* bm2 = (const float*)d_in[23];
  const float* Wm3 = (const float*)d_in[24];
  const float* bm3 = (const float*)d_in[25];

  // Workspace layout (bytes)
  char* ws = (char*)d_ws;
  const size_t FEAT = (size_t)N_NODES * D_H * sizeof(float);   // 2 MiB
  float*    xl     = (float*)(ws + 0 * FEAT);
  float*    xr     = (float*)(ws + 1 * FEAT);
  float*    hA     = (float*)(ws + 2 * FEAT);
  float*    hB     = (float*)(ws + 3 * FEAT);
  float*    agg    = (float*)(ws + 4 * FEAT);
  char*     p      = ws + 5 * FEAT;
  float*    scores = (float*)p;            p += (size_t)N_ITEMS * sizeof(float);
  float*    expa   = (float*)p;            p += (size_t)N_ITEMS * sizeof(float);
  unsigned* mbits  = (unsigned*)p;         p += (size_t)N_NODES * sizeof(unsigned);
  float*    denom  = (float*)p;            p += (size_t)N_NODES * sizeof(float);
  float*    h1acc  = (float*)p;            p += (size_t)D_H * sizeof(float);
  float*    h2     = (float*)p;

  // --- three GATv2 layers ---
  run_gat_layer(x,  D_IN, ei,
                (const float*)d_in[2],  (const float*)d_in[3],
                (const float*)d_in[4],  (const float*)d_in[5],
                (const float*)d_in[6],  (const float*)d_in[7],
                xl, xr, agg, scores, expa, mbits, denom, hA, stream);
  run_gat_layer(hA, D_H,  ei,
                (const float*)d_in[8],  (const float*)d_in[9],
                (const float*)d_in[10], (const float*)d_in[11],
                (const float*)d_in[12], (const float*)d_in[13],
                xl, xr, agg, scores, expa, mbits, denom, hB, stream);
  run_gat_layer(hB, D_H,  ei,
                (const float*)d_in[14], (const float*)d_in[15],
                (const float*)d_in[16], (const float*)d_in[17],
                (const float*)d_in[18], (const float*)d_in[19],
                xl, xr, agg, scores, expa, mbits, denom, hA, stream);

  // --- MLP head: v = hA flattened (524288), Wm1 is 1 GiB -> streaming matvec ---
  hipMemsetAsync(h1acc, 0, D_H * sizeof(float), stream);
  mlp1_partial<<<(N_NODES * D_H) / 256, 256, 0, stream>>>(hA, Wm1, h1acc);
  mlp2<<<2, 256, 0, stream>>>(h1acc, bm1, Wm2, bm2, h2);
  mlp3<<<1, 256, 0, stream>>>(h2, Wm3, bm3, (float*)d_out);
}
